// Severity_66090956751046
// MI455X (gfx1250) — compile-verified, matrix operand order checked by reading, not run on hardware
//
#include <hip/hip_runtime.h>
#include <math.h>

typedef unsigned short u16;
typedef __attribute__((ext_vector_type(16))) __bf16    v16bf;
typedef __attribute__((ext_vector_type(8)))  float     v8f;
typedef __attribute__((ext_vector_type(4)))  unsigned  v4u;
typedef __attribute__((ext_vector_type(8)))  int       v8i;
typedef __attribute__((ext_vector_type(4)))  int       v4i;

__device__ __forceinline__ float bf2f(u16 h) {
    unsigned u = ((unsigned)h) << 16; float f; __builtin_memcpy(&f, &u, 4); return f;
}
__device__ __forceinline__ u16 f2bf(float f) {
    unsigned u; __builtin_memcpy(&u, &f, 4);
    u += 0x7FFFu + ((u >> 16) & 1u);          // round-to-nearest-even
    return (u16)(u >> 16);
}

// ---------------------------------------------------------------------------
// f32 -> bf16 conversion (plain, for deconv weights + input)
// ---------------------------------------------------------------------------
__global__ void cvt_f32_bf16(const float* __restrict__ in, u16* __restrict__ out, long n) {
    long i  = (long)blockIdx.x * blockDim.x + threadIdx.x;
    long st = (long)gridDim.x * blockDim.x;
    for (; i < n; i += st) out[i] = f2bf(in[i]);
}

// f32 [Cout][Cin][3][3]  ->  bf16 [9][Cout][Cin]  (TDM/WMMA-friendly layout:
// per-tap weight plane has contiguous Cin rows, stride Cin)
__global__ void repack_w_bf16(const float* __restrict__ in, u16* __restrict__ out,
                              int Cout, int Cin) {
    const long plane = (long)Cout * Cin;
    const long total = plane * 9;
    const long st    = (long)gridDim.x * blockDim.x;
    for (long i = (long)blockIdx.x * blockDim.x + threadIdx.x; i < total; i += st) {
        const int tap = (int)(i / plane);
        const long r  = i - (long)tap * plane;
        const int co  = (int)(r / Cin);
        const int ci  = (int)(r - (long)co * Cin);
        out[i] = f2bf(in[((size_t)co * Cin + ci) * 9 + tap]);
    }
}

__global__ void fill_u16(u16* __restrict__ p, long n) {
    long i  = (long)blockIdx.x * blockDim.x + threadIdx.x;
    long st = (long)gridDim.x * blockDim.x;
    for (; i < n; i += st) p[i] = 0;
}

// ---------------------------------------------------------------------------
// WMMA implicit-GEMM 3x3 conv, pad=1, bf16 in / f32 accum / bf16 out.
//   - up to 3 input tensors, virtually concatenated along channels
//   - optional sample remap (GNN gather), optional accumulate mode:
//       out += scale * relu(conv + bias)   (mean over 3 neighbors)
//   - weight tile staged to LDS by the Tensor Data Mover (wave 0 builds a 2D
//     D# descriptor: 32 rows (cout) x 32 contiguous bf16 (cin), stride Cin)
// Block: 256 threads = 8 waves. Block tile: 32 Cout x 128 pixels.
// Wave (cow, pxw) owns 16 Cout x 32 px -> 2 accumulators, 2 WMMA per K-step
// (A fragment reused across both), K = Cin x 9 taps in 32-wide bf16 chunks.
// ---------------------------------------------------------------------------
__global__ void __launch_bounds__(256)
conv3x3_wmma(const u16* __restrict__ inA, int Ca,
             const u16* __restrict__ inB, int Cb,
             const u16* __restrict__ inC, int Cc,
             const u16* __restrict__ wgt_t,   // [9][Cout][Cin] bf16 (repacked)
             const float* __restrict__ bias,  // [Cout] f32
             u16* __restrict__ out,           // [N][Cout][H][W] bf16
             int N, int H, int W, int Cout,
             int do_relu, int acc_mode, float scale,
             const int* __restrict__ remap, int remap_off) {
    const int Cin = Ca + Cb + Cc;
    const int HW  = H * W;
    __shared__ u16 wbuf[32 * 32];    // [co_local][k]   (A staging, TDM dest)
    __shared__ u16 ibuf[128 * 32];   // [px_local][k]   (B staging)

    const int tid  = threadIdx.x;
    const int lane = tid & 31;
    const int wave = tid >> 5;
    const int pxw  = wave & 3;       // pixel sub-tile 0..3 (32 px each)
    const int cow  = wave >> 2;      // cout sub-tile 0..1
    const int px0  = blockIdx.x * 128;
    const int co0  = blockIdx.y * 32;
    const int n    = blockIdx.z;
    const int ns   = remap ? remap[n * 3 + remap_off] : n;

    const int mrow = lane & 15;      // A row / B col / D col for this lane
    const int kh   = lane >> 4;      // lane-half selects k sub-range

    v8f acc[2] = {{}, {}};

    for (int c0 = 0; c0 < Cin; c0 += 32) {
        for (int tap = 0; tap < 9; ++tap) {
            const int ky = tap / 3 - 1;
            const int kx = tap % 3 - 1;

            // ---- weight tile via Tensor Data Mover (one DMA per workgroup) ----
#if __has_builtin(__builtin_amdgcn_tensor_load_to_lds)
            if (wave == 0) {
                const unsigned long long ga =
                    (unsigned long long)(const void*)wgt_t +
                    2ull * (((size_t)tap * Cout + co0) * Cin + c0);
                const unsigned ldsa = (unsigned)(unsigned long long)(void*)wbuf;
                // D# group 0: count=1, lds_addr, global_addr[56:0], type=2
                v4u g0 = { 1u, ldsa,
                           (unsigned)(ga & 0xFFFFFFFFull),
                           (unsigned)((ga >> 32) & 0x01FFFFFFull) | 0x80000000u };
                // D# group 1: data_size=2B; tensor 2D (Cin x Cout), tile 32x32,
                // dim0_stride=Cin; OOB reads return zero (= our padding)
                v8i g1 = { (int)(1u << 16),
                           (int)(((unsigned)Cin & 0xFFFFu) << 16),
                           (int)((((unsigned)Cin >> 16) & 0xFFFFu) |
                                 (((unsigned)Cout & 0xFFFFu) << 16)),
                           (int)((((unsigned)Cout >> 16) & 0xFFFFu) | (32u << 16)),
                           (int)32u,
                           (int)(unsigned)Cin,
                           0, 0 };
                v4i g2 = {0, 0, 0, 0}, g3 = {0, 0, 0, 0};
                v8i g4 = {0, 0, 0, 0, 0, 0, 0, 0};
                __builtin_amdgcn_tensor_load_to_lds(g0, g1, g2, g3, g4, 0);
                __builtin_amdgcn_s_wait_tensorcnt(0);
            }
#else
            for (int i = tid; i < 32 * 32; i += 256) {
                const int col = i >> 5, k = i & 31;
                const int ci = c0 + k, co = co0 + col;
                u16 v = 0;
                if (ci < Cin && co < Cout)
                    v = wgt_t[((size_t)tap * Cout + co) * Cin + ci];
                wbuf[(col << 5) | k] = v;
            }
#endif
            __builtin_prefetch(&wgt_t[((size_t)tap * Cout + co0) * Cin + c0 + 32], 0, 1);

            // ---- input slice: 128 pixels x 32 cin, tap-shifted, zero-padded ----
            for (int i = tid; i < 128 * 32; i += 256) {
                const int pl = i >> 5, k = i & 31;
                const int ci = c0 + k;
                const int px = px0 + pl;
                u16 v = 0;
                if (ci < Cin && px < HW) {
                    const int y = px / W, x = px - y * W;
                    const int iy = y + ky, ix = x + kx;
                    if (iy >= 0 && iy < H && ix >= 0 && ix < W) {
                        if (ci < Ca)
                            v = inA[(((size_t)ns * Ca + ci) * H + iy) * W + ix];
                        else if (ci < Ca + Cb)
                            v = inB[(((size_t)ns * Cb + (ci - Ca)) * H + iy) * W + ix];
                        else
                            v = inC[(((size_t)ns * Cc + (ci - Ca - Cb)) * H + iy) * W + ix];
                    }
                }
                ibuf[(pl << 5) | k] = v;
            }
            __syncthreads();

            // Fragments per ISA 16-bit A/B layout: lane<16 k in {0..7,16..23},
            // lane>=16 k in {8..15,24..31}; dword-packed pairs -> b128 LDS reads.
            union { unsigned u[8]; v16bf v; } afr, bfr0, bfr1;
            const unsigned* wrow  = (const unsigned*)&wbuf[((cow << 4) | mrow) << 5];
            const unsigned* irow0 = (const unsigned*)&ibuf[((pxw << 5) | mrow) << 5];
            const unsigned* irow1 = (const unsigned*)&ibuf[((pxw << 5) | 16 | mrow) << 5];
#pragma unroll
            for (int r = 0; r < 4; ++r) {
                afr.u[r]      = wrow[(kh << 2) + r];
                afr.u[4 + r]  = wrow[8 + (kh << 2) + r];
                bfr0.u[r]     = irow0[(kh << 2) + r];
                bfr0.u[4 + r] = irow0[8 + (kh << 2) + r];
                bfr1.u[r]     = irow1[(kh << 2) + r];
                bfr1.u[4 + r] = irow1[8 + (kh << 2) + r];
            }
            acc[0] = __builtin_amdgcn_wmma_f32_16x16x32_bf16(
                false, afr.v, false, bfr0.v, (short)0, acc[0], false, false);
            acc[1] = __builtin_amdgcn_wmma_f32_16x16x32_bf16(
                false, afr.v, false, bfr1.v, (short)0, acc[1], false, false);
            __syncthreads();
        }
    }

    // Epilogue: D layout -> lane col = lane&15, row = r + 8*(lane>>4)
#pragma unroll
    for (int sub = 0; sub < 2; ++sub) {
#pragma unroll
        for (int r = 0; r < 8; ++r) {
            const int mo = r + (kh << 3);
            const int co = co0 + (cow << 4) + mo;
            const int px = px0 + (pxw << 5) + (sub << 4) + mrow;
            if (co < Cout && px < HW) {
                float v = acc[sub][r] + bias[co];
                if (do_relu) v = fmaxf(v, 0.f);
                const size_t o = ((size_t)n * Cout + co) * HW + px;
                if (acc_mode) out[o] = f2bf(bf2f(out[o]) + scale * v);
                else          out[o] = f2bf(v);
            }
        }
    }
}

// ---------------------------------------------------------------------------
// ConvTranspose2d(k=3, s=2, p=1, op=1) == conv on 2x-dilated input, pad(1,2).
// Output pixel touches <=4 parity-valid taps. relu fused. (orig OIHW layout)
// ---------------------------------------------------------------------------
__global__ void deconv2x_kernel(const u16* __restrict__ in,
                                const u16* __restrict__ wgt,   // [Cout][Cin][3][3]
                                const float* __restrict__ bias,
                                u16* __restrict__ out,
                                int N, int Cin, int Cout, int H, int W) {
    const int OH = 2 * H, OW = 2 * W;
    const long total  = (long)N * Cout * OH * OW;
    const long stride = (long)gridDim.x * blockDim.x;
    for (long i = (long)blockIdx.x * blockDim.x + threadIdx.x; i < total; i += stride) {
        const int ox = (int)(i % OW);
        const int oy = (int)((i / OW) % OH);
        const int co = (int)((i / ((long)OW * OH)) % Cout);
        const int n  = (int)(i / ((long)OW * OH * Cout));
        float acc = bias[co];
        for (int ky = 0; ky < 3; ++ky) {
            const int t = oy + ky - 1;
            if (t < 0 || t > 2 * H - 2 || (t & 1)) continue;
            const int iy = t >> 1;
            for (int kx = 0; kx < 3; ++kx) {
                const int s = ox + kx - 1;
                if (s < 0 || s > 2 * W - 2 || (s & 1)) continue;
                const int ix = s >> 1;
                const u16* ip = in + ((size_t)n * Cin * H + iy) * W + ix;
                const u16* wp = wgt + (size_t)co * Cin * 9 + (ky * 3 + kx);
                for (int ci = 0; ci < Cin; ++ci)
                    acc += bf2f(ip[(size_t)ci * H * W]) * bf2f(wp[(size_t)ci * 9]);
            }
        }
        out[i] = f2bf(fmaxf(acc, 0.f));
    }
}

// ---------------------------------------------------------------------------
__global__ void maxpool2_kernel(const u16* __restrict__ in, u16* __restrict__ out,
                                int N, int C, int H, int W) {
    const int OH = H >> 1, OW = W >> 1;
    const long total  = (long)N * C * OH * OW;
    const long stride = (long)gridDim.x * blockDim.x;
    for (long i = (long)blockIdx.x * blockDim.x + threadIdx.x; i < total; i += stride) {
        const int ox = (int)(i % OW);
        const int oy = (int)((i / OW) % OH);
        const int c  = (int)((i / ((long)OW * OH)) % C);
        const int n  = (int)(i / ((long)OW * OH * C));
        const u16* p = in + (((size_t)n * C + c) * H + oy * 2) * W + ox * 2;
        float m = fmaxf(fmaxf(bf2f(p[0]), bf2f(p[1])),
                        fmaxf(bf2f(p[W]), bf2f(p[W + 1])));
        out[i] = f2bf(m);
    }
}

// ---------------------------------------------------------------------------
// Gram matrix over 8 graph nodes; node features = (a[Da] ++ b[Db]) flat bf16.
// ---------------------------------------------------------------------------
__global__ void gram_kernel(const u16* __restrict__ a, long Da,
                            const u16* __restrict__ b, long Db,
                            float* __restrict__ gram) {
    __shared__ float red[256];
    const int i = blockIdx.x >> 3, j = blockIdx.x & 7;
    float s = 0.f;
    for (long d = threadIdx.x; d < Da; d += 256)
        s += bf2f(a[(size_t)i * Da + d]) * bf2f(a[(size_t)j * Da + d]);
    if (b)
        for (long d = threadIdx.x; d < Db; d += 256)
            s += bf2f(b[(size_t)i * Db + d]) * bf2f(b[(size_t)j * Db + d]);
    red[threadIdx.x] = s;
    __syncthreads();
    for (int o = 128; o > 0; o >>= 1) {
        if (threadIdx.x < o) red[threadIdx.x] += red[threadIdx.x + o];
        __syncthreads();
    }
    if (threadIdx.x == 0) gram[i * 8 + j] = red[0];
}

// cosine-sim top-3 per node, self excluded
__global__ void topk3_kernel(const float* __restrict__ gram, int* __restrict__ idx) {
    const int n = threadIdx.x;
    if (n >= 8) return;
    float nrm[8], sim[8];
    for (int j = 0; j < 8; ++j) nrm[j] = fmaxf(sqrtf(fmaxf(gram[j * 8 + j], 0.f)), 1e-8f);
    for (int j = 0; j < 8; ++j)
        sim[j] = (j == n) ? -1e30f : gram[n * 8 + j] / (nrm[n] * nrm[j]);
    for (int k = 0; k < 3; ++k) {
        float best = -1e31f; int bi = 0;
        for (int j = 0; j < 8; ++j)
            if (sim[j] > best) { best = sim[j]; bi = j; }
        sim[bi] = -1e30f;
        idx[n * 3 + k] = bi;
    }
}

// final 1x1 conv, 32 -> 2 channels, f32 weights/output
__global__ void conv1x1_out_kernel(const u16* __restrict__ x,
                                   const float* __restrict__ w,  // [2][32]
                                   const float* __restrict__ b,
                                   float* __restrict__ out, int N, int HW) {
    const long total  = (long)N * 2 * HW;
    const long stride = (long)gridDim.x * blockDim.x;
    for (long i = (long)blockIdx.x * blockDim.x + threadIdx.x; i < total; i += stride) {
        const int px = (int)(i % HW);
        const int co = (int)((i / HW) % 2);
        const int n  = (int)(i / (2L * HW));
        float acc = b[co];
        for (int ci = 0; ci < 32; ++ci)
            acc += bf2f(x[((size_t)n * 32 + ci) * HW + px]) * w[co * 32 + ci];
        out[i] = acc;
    }
}

// ---------------------------------------------------------------------------
// Host orchestration
// ---------------------------------------------------------------------------
static void launch_conv(hipStream_t s,
                        const u16* A, int Ca, const u16* B, int Cb, const u16* C, int Cc,
                        const u16* w, const float* bias, u16* out,
                        int N, int H, int W, int Cout,
                        int relu, int accm, float scale,
                        const int* remap, int roff) {
    const int HW = H * W;
    dim3 g((HW + 127) / 128, (Cout + 31) / 32, N);
    conv3x3_wmma<<<g, 256, 0, s>>>(A, Ca, B, Cb, C, Cc, w, bias, out,
                                   N, H, W, Cout, relu, accm, scale, remap, roff);
}

static void launch_flat(hipStream_t s, long total, int& blocks) {
    blocks = (int)((total + 255) / 256);
    if (blocks > 65535) blocks = 65535;
}

extern "C" void kernel_launch(void* const* d_in, const int* in_sizes, int n_in,
                              void* d_out, int out_size, void* d_ws, size_t ws_size,
                              hipStream_t stream) {
    (void)in_sizes; (void)n_in; (void)out_size; (void)ws_size;
    const int N = 8, F0 = 192;
    const long HW0 = (long)F0 * F0;

    char* base = (char*)d_ws;
    size_t off = 0;
    auto alloc = [&](size_t bytes) -> void* {
        off = (off + 255) & ~(size_t)255;
        void* p = base + off;
        off += bytes;
        return p;
    };

    // -------- convert weights to bf16; 3x3-conv weights repacked [9][Co][Ci] --------
    const int  widx[24] = {1,3,5,7,9,11,13,15,17,19,21,23,25,27,29,31,33,35,37,39,41,43,45,47};
    const int  wco[24]  = {32,32,64,64,128,128,256,256,512,512,256,256,256,128,128,128,64,64,64,32,64,32,32,32};
    const int  wci[24]  = {1,32,32,64,64,128,128,256,256,512,512,512,256,256,256,128,128,128,64,64,64,128,32,64};
    const int  wdec[24] = {0,0,0,0,0,0,0,0,0,0,1,0,0,1,0,0,1,0,0,1,0,0,0,0};
    u16* wb[24];
    for (int i = 0; i < 24; ++i) {
        const long n = (long)wco[i] * wci[i] * 9;
        wb[i] = (u16*)alloc((size_t)n * 2);
        int g; launch_flat(stream, n, g);
        if (wdec[i])
            cvt_f32_bf16<<<g, 256, 0, stream>>>((const float*)d_in[widx[i]], wb[i], n);
        else
            repack_w_bf16<<<g, 256, 0, stream>>>((const float*)d_in[widx[i]], wb[i],
                                                 wco[i], wci[i]);
    }
    auto bias = [&](int i) { return (const float*)d_in[i]; };

    // -------- activation buffers (bf16) --------
    u16* in_bf = (u16*)alloc(2 * (size_t)N * HW0);
    u16* tmp   = (u16*)alloc(2 * (size_t)N * 32 * HW0);   // shared double-conv temp
    u16* d1  = (u16*)alloc(2 * (size_t)N * 32 * HW0);
    u16* p1  = (u16*)alloc(2 * (size_t)N * 32 * 96 * 96);
    u16* d2  = (u16*)alloc(2 * (size_t)N * 64 * 96 * 96);
    u16* p2  = (u16*)alloc(2 * (size_t)N * 64 * 48 * 48);
    u16* d3  = (u16*)alloc(2 * (size_t)N * 128 * 48 * 48);
    u16* p3  = (u16*)alloc(2 * (size_t)N * 128 * 24 * 24);
    u16* d4  = (u16*)alloc(2 * (size_t)N * 256 * 24 * 24);
    u16* p4  = (u16*)alloc(2 * (size_t)N * 256 * 12 * 12);
    u16* d5  = (u16*)alloc(2 * (size_t)N * 512 * 12 * 12);
    u16* t1  = (u16*)alloc(2 * (size_t)N * 256 * 24 * 24);
    u16* u2  = (u16*)alloc(2 * (size_t)N * 256 * 24 * 24);
    u16* t2  = (u16*)alloc(2 * (size_t)N * 128 * 48 * 48);
    u16* u3  = (u16*)alloc(2 * (size_t)N * 128 * 48 * 48);
    u16* t3  = (u16*)alloc(2 * (size_t)N * 64 * 96 * 96);
    u16* u4  = (u16*)alloc(2 * (size_t)N * 64 * 96 * 96);
    u16* t4  = (u16*)alloc(2 * (size_t)N * 32 * HW0);
    u16* ag1 = (u16*)alloc(2 * (size_t)N * 64 * HW0);
    u16* g1  = (u16*)alloc(2 * (size_t)N * 32 * HW0);
    u16* ag2 = (u16*)alloc(2 * (size_t)N * 32 * HW0);
    u16* g2  = (u16*)alloc(2 * (size_t)N * 32 * HW0);
    float* gram = (float*)alloc(64 * sizeof(float));
    int*   nidx = (int*)alloc(24 * sizeof(int));

    int g;
    launch_flat(stream, N * HW0, g);
    cvt_f32_bf16<<<g, 256, 0, stream>>>((const float*)d_in[0], in_bf, N * HW0);

    // -------- encoder --------
    launch_conv(stream, in_bf, 1, 0, 0, 0, 0, wb[0], bias(2), tmp, N, 192, 192, 32, 1, 0, 0, 0, 0);
    launch_conv(stream, tmp, 32, 0, 0, 0, 0, wb[1], bias(4), d1,  N, 192, 192, 32, 1, 0, 0, 0, 0);
    launch_flat(stream, (long)N * 32 * 96 * 96, g);
    maxpool2_kernel<<<g, 256, 0, stream>>>(d1, p1, N, 32, 192, 192);

    launch_conv(stream, p1, 32, 0, 0, 0, 0, wb[2], bias(6), tmp, N, 96, 96, 64, 1, 0, 0, 0, 0);
    launch_conv(stream, tmp, 64, 0, 0, 0, 0, wb[3], bias(8), d2,  N, 96, 96, 64, 1, 0, 0, 0, 0);
    launch_flat(stream, (long)N * 64 * 48 * 48, g);
    maxpool2_kernel<<<g, 256, 0, stream>>>(d2, p2, N, 64, 96, 96);

    launch_conv(stream, p2, 64, 0, 0, 0, 0, wb[4], bias(10), tmp, N, 48, 48, 128, 1, 0, 0, 0, 0);
    launch_conv(stream, tmp, 128, 0, 0, 0, 0, wb[5], bias(12), d3, N, 48, 48, 128, 1, 0, 0, 0, 0);
    launch_flat(stream, (long)N * 128 * 24 * 24, g);
    maxpool2_kernel<<<g, 256, 0, stream>>>(d3, p3, N, 128, 48, 48);

    launch_conv(stream, p3, 128, 0, 0, 0, 0, wb[6], bias(14), tmp, N, 24, 24, 256, 1, 0, 0, 0, 0);
    launch_conv(stream, tmp, 256, 0, 0, 0, 0, wb[7], bias(16), d4, N, 24, 24, 256, 1, 0, 0, 0, 0);
    launch_flat(stream, (long)N * 256 * 12 * 12, g);
    maxpool2_kernel<<<g, 256, 0, stream>>>(d4, p4, N, 256, 24, 24);

    launch_conv(stream, p4, 256, 0, 0, 0, 0, wb[8], bias(18), tmp, N, 12, 12, 512, 1, 0, 0, 0, 0);
    launch_conv(stream, tmp, 512, 0, 0, 0, 0, wb[9], bias(20), d5, N, 12, 12, 512, 1, 0, 0, 0, 0);

    // -------- decoder (concats are virtual: two-pointer conv inputs) --------
    launch_flat(stream, (long)N * 256 * 24 * 24, g);
    deconv2x_kernel<<<g, 256, 0, stream>>>(d5, wb[10], bias(22), t1, N, 512, 256, 12, 12);
    launch_conv(stream, d4, 256, t1, 256, 0, 0, wb[11], bias(24), tmp, N, 24, 24, 256, 1, 0, 0, 0, 0);
    launch_conv(stream, tmp, 256, 0, 0, 0, 0, wb[12], bias(26), u2,  N, 24, 24, 256, 1, 0, 0, 0, 0);

    launch_flat(stream, (long)N * 128 * 48 * 48, g);
    deconv2x_kernel<<<g, 256, 0, stream>>>(u2, wb[13], bias(28), t2, N, 256, 128, 24, 24);
    launch_conv(stream, d3, 128, t2, 128, 0, 0, wb[14], bias(30), tmp, N, 48, 48, 128, 1, 0, 0, 0, 0);
    launch_conv(stream, tmp, 128, 0, 0, 0, 0, wb[15], bias(32), u3,  N, 48, 48, 128, 1, 0, 0, 0, 0);

    launch_flat(stream, (long)N * 64 * 96 * 96, g);
    deconv2x_kernel<<<g, 256, 0, stream>>>(u3, wb[16], bias(34), t3, N, 128, 64, 48, 48);
    launch_conv(stream, d2, 64, t3, 64, 0, 0, wb[17], bias(36), tmp, N, 96, 96, 64, 1, 0, 0, 0, 0);
    launch_conv(stream, tmp, 64, 0, 0, 0, 0, wb[18], bias(38), u4,  N, 96, 96, 64, 1, 0, 0, 0, 0);

    launch_flat(stream, (long)N * 32 * HW0, g);
    deconv2x_kernel<<<g, 256, 0, stream>>>(u4, wb[19], bias(40), t4, N, 64, 32, 96, 96);

    // -------- GNN block 1: x = concat(d1, t4) [virtual] --------
    gram_kernel<<<64, 256, 0, stream>>>(d1, 32 * HW0, t4, 32 * HW0, gram);
    topk3_kernel<<<1, 32, 0, stream>>>(gram, nidx);
    launch_flat(stream, (long)N * 64 * HW0, g);
    fill_u16<<<g, 256, 0, stream>>>(ag1, (long)N * 64 * HW0);
    for (int k = 0; k < 3; ++k)   // agg = mean_k relu(conv(x[idx[:,k]]))
        launch_conv(stream, d1, 32, t4, 32, 0, 0, wb[20], bias(42), ag1,
                    N, 192, 192, 64, 1, 1, 1.f / 3.f, nidx, k);
    launch_conv(stream, d1, 32, t4, 32, ag1, 64, wb[21], bias(44), g1,
                N, 192, 192, 32, 1, 0, 0, 0, 0);

    // -------- GNN block 2 --------
    gram_kernel<<<64, 256, 0, stream>>>(g1, 32 * HW0, (const u16*)0, 0, gram);
    topk3_kernel<<<1, 32, 0, stream>>>(gram, nidx);
    launch_flat(stream, (long)N * 32 * HW0, g);
    fill_u16<<<g, 256, 0, stream>>>(ag2, (long)N * 32 * HW0);
    for (int k = 0; k < 3; ++k)
        launch_conv(stream, g1, 32, 0, 0, 0, 0, wb[22], bias(46), ag2,
                    N, 192, 192, 32, 1, 1, 1.f / 3.f, nidx, k);
    launch_conv(stream, g1, 32, ag2, 32, 0, 0, wb[23], bias(48), g2,
                N, 192, 192, 32, 1, 0, 0, 0, 0);

    // -------- final 1x1 conv -> f32 output --------
    launch_flat(stream, (long)N * 2 * HW0, g);
    conv1x1_out_kernel<<<g, 256, 0, stream>>>(g2, (const float*)d_in[49],
                                              (const float*)d_in[50],
                                              (float*)d_out, N, (int)HW0);
}
